// GATModel_65515431133471
// MI455X (gfx1250) — matrix operand order repaired and checked by reading, not compiled, not run on hardware
//
#include <hip/hip_runtime.h>
#include <math.h>

// ---------------------------------------------------------------------------
// GAT (2-layer) for MI455X / gfx1250.
// GEMMs: v_wmma_f32_16x16x32_bf16, K=128 fully unrolled, B pre-packed into
// fragment layout (b128 loads), 5 M-tiles per wave reusing register B frags.
// Edge softmax / aggregation: edge-parallel kernels, L2-resident f32 atomics.
// ---------------------------------------------------------------------------

typedef __bf16 bf16;
typedef bf16  bf16x8  __attribute__((ext_vector_type(8)));
typedef bf16  bf16x16 __attribute__((ext_vector_type(16)));
typedef float f32x8   __attribute__((ext_vector_type(8)));
typedef float f32x4   __attribute__((ext_vector_type(4)));

#define H1 8
#define C1 16
#define F1 128   // H1*C1
#define C2 16
#define KDIM 128 // K for both GEMMs
#define TMW 5    // M tiles per wave (50000/16 = 3125 = 625*5)

// ---- monotone f32 <-> u32 key (atomicMax on floats, any sign) --------------
__device__ __forceinline__ unsigned enc_f32(float f) {
  unsigned u = __float_as_uint(f);
  return (u & 0x80000000u) ? ~u : (u | 0x80000000u);
}
__device__ __forceinline__ float dec_f32(unsigned k) {
  unsigned u = (k & 0x80000000u) ? (k ^ 0x80000000u) : ~k;
  return __uint_as_float(u);
}

__device__ __forceinline__ float leaky02(float e) { return e > 0.0f ? e : 0.2f * e; }

// ---------------------------------------------------------------------------
__global__ void k_f32_to_bf16(const float* __restrict__ in, bf16* __restrict__ out,
                              long long n) {
  long long i = blockIdx.x * (long long)blockDim.x + threadIdx.x;
  if (i < n) out[i] = (bf16)in[i];
}

__global__ void k_zero_f32(float* __restrict__ p, long long n) {
  long long i = blockIdx.x * (long long)blockDim.x + threadIdx.x;
  if (i < n) p[i] = 0.0f;
}

__global__ void k_init_bias(float* __restrict__ out, const float* __restrict__ b,
                            long long n, int F) {
  long long i = blockIdx.x * (long long)blockDim.x + threadIdx.x;
  if (i < n) out[i] = b[i % F];
}

// ---------------------------------------------------------------------------
// Pack f32 W[K,N] (row-major) into bf16 WMMA B-fragments:
//   frag f = tileN*(K/32) + ks : 32 lanes x 16 bf16, contiguous per lane.
//   lane L holds column n = tileN*16 + (L&15); elem e -> k = ks*32 + 16*(L>>4) + e
__global__ void k_pack_B(const float* __restrict__ W, bf16* __restrict__ Bp,
                         int N, int K) {
  int t = blockIdx.x * blockDim.x + threadIdx.x;
  int total = (N >> 4) * (K >> 5) * 512;
  if (t >= total) return;
  int e     = t & 15;
  int lane  = (t >> 4) & 31;
  int frag  = t >> 9;
  int ksb   = K >> 5;
  int ks    = frag % ksb;
  int tileN = frag / ksb;
  int n = (tileN << 4) + (lane & 15);
  int k = (ks << 5) + ((lane >> 4) << 4) + e;
  Bp[t] = (bf16)W[(size_t)k * N + n];
}

// ---------------------------------------------------------------------------
// C[M,N] = A[M,128] x B[128,N].  One wave per (group-of-5 M tiles, N tile).
// B fragments (4 x 16 bf16 per lane) loaded once as b128 pairs and held in
// registers across all 5 M tiles. A fragments: 2 x b128 per WMMA.
// EXEC stays all-ones: out-of-range waves/tiles clamp and skip only stores.
// ---------------------------------------------------------------------------
__global__ void k_wmma_gemm_bf16_k128(const bf16* __restrict__ A,
                                      const bf16* __restrict__ Bp,
                                      float* __restrict__ C, int M, int N) {
  const int lane = threadIdx.x & 31;
  int gwave   = (int)((blockIdx.x * (long long)blockDim.x + threadIdx.x) >> 5);
  int tilesM  = M >> 4;
  int tilesN  = N >> 4;
  int groupsM = (tilesM + TMW - 1) / TMW;
  int ngroups = groupsM * tilesN;
  bool valid  = gwave < ngroups;
  int g       = valid ? gwave : 0;
  int tileN   = g % tilesN;
  int gM      = g / tilesN;
  int hf      = lane >> 4;
  int l15     = lane & 15;

  // ---- load 4 B fragments (whole K=128) once ----
  bf16x16 bfrag[4];
#pragma unroll
  for (int ks = 0; ks < 4; ++ks) {
    const bf16* bp = Bp + (((size_t)(tileN * 4 + ks) * 32) + lane) * 16;
    bf16x8 lo = *(const bf16x8*)bp;
    bf16x8 hi = *(const bf16x8*)(bp + 8);
    bfrag[ks] = __builtin_shufflevector(lo, hi, 0, 1, 2, 3, 4, 5, 6, 7,
                                        8, 9, 10, 11, 12, 13, 14, 15);
  }

#pragma unroll
  for (int mt = 0; mt < TMW; ++mt) {
    int tileMr = gM * TMW + mt;
    bool tvalid = valid && (tileMr < tilesM);
    int tileM = tileMr < tilesM ? tileMr : (tilesM - 1);  // clamp for compute
    int m = (tileM << 4) + l15;

    f32x8 acc = {0.f, 0.f, 0.f, 0.f, 0.f, 0.f, 0.f, 0.f};
#pragma unroll
    for (int ks = 0; ks < 4; ++ks) {
      const bf16* ap = A + (size_t)m * KDIM + ks * 32 + 16 * hf;
      bf16x8 lo = *(const bf16x8*)ap;          // K = ks*32 + 16*hf + [0..7]
      bf16x8 hi = *(const bf16x8*)(ap + 8);    // K = ks*32 + 16*hf + 8 + [0..7]
      bf16x16 a = __builtin_shufflevector(lo, hi, 0, 1, 2, 3, 4, 5, 6, 7,
                                          8, 9, 10, 11, 12, 13, 14, 15);
      acc = __builtin_amdgcn_wmma_f32_16x16x32_bf16(
          false, a, false, bfrag[ks], (short)0, acc, false, false);
    }
    if (tvalid) {
#pragma unroll
      for (int v = 0; v < 8; ++v)
        C[(size_t)((tileM << 4) + v + 8 * hf) * N + (tileN << 4) + l15] = acc[v];
    }
  }
}

// ---------------------------------------------------------------------------
// Per (node, head): alpha_s = <h, a_src>, alpha_d = <h, a_dst>, init max/denom.
__global__ void k_node_prep(const float* __restrict__ h,
                            const float* __restrict__ a_src, const float* __restrict__ a_dst,
                            float* __restrict__ alpha_s, float* __restrict__ alpha_d,
                            unsigned* __restrict__ mkey, float* __restrict__ denom,
                            int N, int H, int C) {
  int t = blockIdx.x * blockDim.x + threadIdx.x;
  if (t >= N * H) return;
  int node = t / H, hd = t % H;
  const float* hp = h + (size_t)node * H * C + (size_t)hd * C;
  float s = 0.f, d = 0.f;
  for (int c = 0; c < C; ++c) {
    float v = hp[c];
    s += v * a_src[hd * C + c];
    d += v * a_dst[hd * C + c];
  }
  alpha_s[t] = s;
  alpha_d[t] = d;
  mkey[t]    = 0u;   // key space minimum
  denom[t]   = 0.f;
}

__device__ __forceinline__ void edge_nodes(const int* __restrict__ ei, long long E,
                                           long long t, int& src, int& dst) {
  if (t < E) { src = ei[t]; dst = ei[E + t]; }
  else       { src = dst = (int)(t - E); }      // self-loops appended
}

// Pass 1: segment max of leaky-relu logits (uint-key atomicMax -> L2 atomic).
template <int H>
__global__ void k_edge_max(const int* __restrict__ ei, long long E, int N,
                           const float* __restrict__ alpha_s,
                           const float* __restrict__ alpha_d,
                           unsigned* __restrict__ mkey) {
  long long t = blockIdx.x * (long long)blockDim.x + threadIdx.x;
  long long Et = E + N;
  if (t >= Et) return;
  int src, dst;
  edge_nodes(ei, E, t, src, dst);
#pragma unroll
  for (int hd = 0; hd < H; ++hd) {
    float e = leaky02(alpha_s[src * H + hd] + alpha_d[dst * H + hd]);
    atomicMax(&mkey[dst * H + hd], enc_f32(e));
  }
}

// Pass 2: denom = segment_sum(exp(e - max)).
template <int H>
__global__ void k_edge_expsum(const int* __restrict__ ei, long long E, int N,
                              const float* __restrict__ alpha_s,
                              const float* __restrict__ alpha_d,
                              const unsigned* __restrict__ mkey,
                              float* __restrict__ denom) {
  long long t = blockIdx.x * (long long)blockDim.x + threadIdx.x;
  long long Et = E + N;
  if (t >= Et) return;
  int src, dst;
  edge_nodes(ei, E, t, src, dst);
#pragma unroll
  for (int hd = 0; hd < H; ++hd) {
    float e  = leaky02(alpha_s[src * H + hd] + alpha_d[dst * H + hd]);
    float ex = __expf(e - dec_f32(mkey[dst * H + hd]));
    atomicAdd(&denom[dst * H + hd], ex);
  }
}

// Pass 3: out[dst] += h[src] * alpha.  Thread per (edge, head); logit
// recomputed (2 small L2 gathers) instead of a 53MB per-edge buffer.
template <int H, int C>
__global__ void k_edge_agg(const int* __restrict__ ei, long long E, int N,
                           const float* __restrict__ hfeat,
                           const float* __restrict__ alpha_s,
                           const float* __restrict__ alpha_d,
                           const unsigned* __restrict__ mkey,
                           const float* __restrict__ denom,
                           float* __restrict__ out) {
  long long t   = blockIdx.x * (long long)blockDim.x + threadIdx.x;
  long long tot = (E + N) * (long long)H;
  if (t >= tot) return;
  long long eidx = t / H;
  int hd = (int)(t % H);
  int src, dst;
  edge_nodes(ei, E, eidx, src, dst);
  float e = leaky02(alpha_s[src * H + hd] + alpha_d[dst * H + hd]);
  float a = __expf(e - dec_f32(mkey[dst * H + hd])) / denom[dst * H + hd];
  const float* hp = hfeat + (size_t)src * H * C + (size_t)hd * C;
  float*       op = out   + (size_t)dst * H * C + (size_t)hd * C;
#pragma unroll
  for (int c4 = 0; c4 < C / 4; ++c4) {
    f32x4 v = *(const f32x4*)(hp + 4 * c4);   // b128 gather
    atomicAdd(&op[4 * c4 + 0], v.x * a);
    atomicAdd(&op[4 * c4 + 1], v.y * a);
    atomicAdd(&op[4 * c4 + 2], v.z * a);
    atomicAdd(&op[4 * c4 + 3], v.w * a);
  }
}

// elu(x + b) and convert to bf16 (feeds GEMM2).
__global__ void k_elu_bias_bf16(const float* __restrict__ in, const float* __restrict__ b,
                                bf16* __restrict__ out, long long n, int F) {
  long long i = blockIdx.x * (long long)blockDim.x + threadIdx.x;
  if (i >= n) return;
  float v = in[i] + b[i % F];
  v = v > 0.0f ? v : (__expf(v) - 1.0f);
  out[i] = (bf16)v;
}

// ---------------------------------------------------------------------------
extern "C" void kernel_launch(void* const* d_in, const int* in_sizes, int n_in,
                              void* d_out, int out_size, void* d_ws, size_t ws_size,
                              hipStream_t stream) {
  const float* x      = (const float*)d_in[0];
  const int*   ei     = (const int*)d_in[1];
  const float* W1     = (const float*)d_in[2];
  const float* a_src1 = (const float*)d_in[3];
  const float* a_dst1 = (const float*)d_in[4];
  const float* b1     = (const float*)d_in[5];
  const float* W2     = (const float*)d_in[6];
  const float* a_src2 = (const float*)d_in[7];
  const float* a_dst2 = (const float*)d_in[8];
  const float* b2     = (const float*)d_in[9];
  float* out = (float*)d_out;

  const int       N  = in_sizes[0] / F1;       // 50000
  const long long E  = in_sizes[1] / 2;        // 1.6M
  const long long Et = E + N;

  // ---- carve workspace (256B aligned) ----
  char* wp = (char*)d_ws;
  auto carve = [&](size_t bytes) -> void* {
    void* r = (void*)wp;
    wp += (bytes + 255) & ~(size_t)255;
    return r;
  };
  bf16*     xb   = (bf16*)    carve((size_t)N * F1 * sizeof(bf16));
  bf16*     W1p  = (bf16*)    carve((size_t)F1 * KDIM * sizeof(bf16));  // packed frags
  bf16*     W2p  = (bf16*)    carve((size_t)C2 * KDIM * sizeof(bf16));  // packed frags
  float*    h1   = (float*)   carve((size_t)N * F1 * sizeof(float));
  float*    as1  = (float*)   carve((size_t)N * H1 * sizeof(float));
  float*    ad1  = (float*)   carve((size_t)N * H1 * sizeof(float));
  unsigned* mk1  = (unsigned*)carve((size_t)N * H1 * sizeof(unsigned));
  float*    dn1  = (float*)   carve((size_t)N * H1 * sizeof(float));
  float*    out1 = (float*)   carve((size_t)N * F1 * sizeof(float));
  bf16*     hab  = (bf16*)    carve((size_t)N * F1 * sizeof(bf16));
  float*    h2   = (float*)   carve((size_t)N * C2 * sizeof(float));
  float*    as2  = (float*)   carve((size_t)N * sizeof(float));
  float*    ad2  = (float*)   carve((size_t)N * sizeof(float));
  unsigned* mk2  = (unsigned*)carve((size_t)N * sizeof(unsigned));
  float*    dn2  = (float*)   carve((size_t)N * sizeof(float));

  const int TB = 256;
  auto blocks1d = [&](long long n) -> unsigned { return (unsigned)((n + TB - 1) / TB); };

  // ---- convert activations, pack weights into WMMA fragment layout ----
  k_f32_to_bf16<<<blocks1d((long long)N * F1), TB, 0, stream>>>(x, xb, (long long)N * F1);
  k_pack_B<<<blocks1d((F1 / 16) * (KDIM / 32) * 512), TB, 0, stream>>>(W1, W1p, F1, KDIM);
  k_pack_B<<<blocks1d((C2 / 16) * (KDIM / 32) * 512), TB, 0, stream>>>(W2, W2p, C2, KDIM);

  // ---- layer 1: h1 = x @ W1 (WMMA) ----
  {
    long long groupsM = ((N / 16) + TMW - 1) / TMW;   // 625
    long long waves   = groupsM * (F1 / 16);
    k_wmma_gemm_bf16_k128<<<blocks1d(waves * 32), TB, 0, stream>>>(xb, W1p, h1, N, F1);
  }
  k_node_prep<<<blocks1d((long long)N * H1), TB, 0, stream>>>(
      h1, a_src1, a_dst1, as1, ad1, mk1, dn1, N, H1, C1);
  k_zero_f32<<<blocks1d((long long)N * F1), TB, 0, stream>>>(out1, (long long)N * F1);

  k_edge_max<H1><<<blocks1d(Et), TB, 0, stream>>>(ei, E, N, as1, ad1, mk1);
  k_edge_expsum<H1><<<blocks1d(Et), TB, 0, stream>>>(ei, E, N, as1, ad1, mk1, dn1);
  k_edge_agg<H1, C1><<<blocks1d(Et * H1), TB, 0, stream>>>(
      ei, E, N, h1, as1, ad1, mk1, dn1, out1);

  // ---- elu(out1 + b1) -> bf16 ----
  k_elu_bias_bf16<<<blocks1d((long long)N * F1), TB, 0, stream>>>(
      out1, b1, hab, (long long)N * F1, F1);

  // ---- layer 2: h2 = h1act @ W2 (WMMA) ----
  {
    long long groupsM = ((N / 16) + TMW - 1) / TMW;
    long long waves   = groupsM * (C2 / 16);
    k_wmma_gemm_bf16_k128<<<blocks1d(waves * 32), TB, 0, stream>>>(hab, W2p, h2, N, C2);
  }
  k_node_prep<<<blocks1d((long long)N), TB, 0, stream>>>(
      h2, a_src2, a_dst2, as2, ad2, mk2, dn2, N, 1, C2);
  k_init_bias<<<blocks1d((long long)N * C2), TB, 0, stream>>>(
      out, b2, (long long)N * C2, C2);

  k_edge_max<1><<<blocks1d(Et), TB, 0, stream>>>(ei, E, N, as2, ad2, mk2);
  k_edge_expsum<1><<<blocks1d(Et), TB, 0, stream>>>(ei, E, N, as2, ad2, mk2, dn2);
  k_edge_agg<1, C2><<<blocks1d(Et), TB, 0, stream>>>(
      ei, E, N, h2, as2, ad2, mk2, dn2, out);
}
